// VQVAE_35802847379921
// MI455X (gfx1250) — matrix operand order repaired and checked by reading, not compiled, not run on hardware
//
#include <hip/hip_runtime.h>
#include <math.h>

typedef __attribute__((ext_vector_type(16))) _Float16 v16h;
typedef __attribute__((ext_vector_type(8)))  _Float16 v8h;
typedef __attribute__((ext_vector_type(8)))  float    v8f;

// ---------------------------------------------------------------------------
// Implicit-GEMM convolution with WMMA f32_16x16x32_f16.
//
// Each wave computes a 32(Cout) x 32(position) tile: 4 WMMAs per K-step
// (2 M-tiles x 2 N-sub-tiles) sharing four A b128 loads. wpb waves per
// block share one LDS-staged pair of B sub-tiles (double-buffered over K).
//
//   M = Cout (f16 weights pre-converted to [Mpad][Kpad], K-major, zero pad)
//   N = B*OH*OW output positions (always a multiple of 32)
//   K = Cin*KH*KW, padded to Kpad (multiple of 32)
//
// B-fragment LDS staging layout (per 32x16 K-chunk, per sub-tile):
//   element (kk, n), kk in [0,32): lane = 16*((kk>>3)&1) + n, elem = (kk&7) + 8*(kk>>4)
//   -> each lane's 16 halves are 32 contiguous bytes: two ds_load_b128.
// A-fragment from f16 weights: lane(m = l16, half) holds
//   k = k0 + 8*half + [0..7]  and  k = k0 + 16 + 8*half + [0..7]
//   -> two contiguous global_load_b128 per 16-row tile.
//
// mode 0: conv(stride,pad). mode 1: ConvTranspose2d(k4,s2,p1) as 2x
// lhs-dilated conv with pad=2. preRelu: relu on input reads. Res: residual
// add source. doTanh: tanh on final store.
// ---------------------------------------------------------------------------
__global__ __launch_bounds__(128)
void conv_igemm_wmma(const float* __restrict__ X, const _Float16* __restrict__ WF,
                     const float* __restrict__ Bias, const float* __restrict__ Res,
                     float* __restrict__ Y,
                     int Cin, int H, int W, int Cout,
                     int OH, int OW, int KH, int KW,
                     int stride, int pad, int mode,
                     int preRelu, int doTanh, int Kpad)
{
    __shared__ __align__(16) _Float16 stage[2][2][512];   // [kbuf][nsub][elem]

    const int tid  = threadIdx.x;
    const int lane = tid & 31;
    const int wv   = tid >> 5;
    const int half = lane >> 4;
    const int l16  = lane & 15;

    const int KHW  = KH * KW;
    const int Ktot = Cin * KHW;
    const int OHW  = OH * OW;
    const int wpb  = blockDim.x >> 5;          // waves per block (divides mtiles)
    const int mtiles  = (Cout + 31) >> 5;
    const int mgroups = mtiles / wpb;

    const int mg = blockIdx.x % mgroups;
    const int nt = blockIdx.x / mgroups;
    const int mbase = (mg * wpb + wv) << 5;
    const int nbase = nt << 5;                 // 32 positions per block

    // Cooperative, branchless im2col gather of one 32x32 K-chunk (two 16-col
    // sub-tiles, 1024 f16 elements) into LDS.
    auto gather_step = [&](int k0, _Float16* base) {
        for (int idx = tid; idx < 1024; idx += blockDim.x) {
            const int sub = idx >> 9;                      // n sub-tile 0/1
            const int gl  = (idx >> 4) & 31;               // dest lane 0..31
            const int e   = idx & 15;                      // elem within lane
            const int k   = k0 + ((gl >> 4) << 3) + (e & 7) + ((e >> 3) << 4);     // K index
            const int gn  = nbase + (sub << 4) + (gl & 15);
            const int gb  = gn / OHW;
            const int gr  = gn - gb * OHW;
            const int goh = gr / OW;
            const int gow = gr - goh * OW;

            const int ci = k / KHW;
            const int r2 = k - ci * KHW;
            const int kh = r2 / KW;
            const int kw = r2 - kh * KW;

            bool ok = (k < Ktot);
            int ih, iw;
            if (mode == 0) {
                ih = goh * stride + kh - pad;
                iw = gow * stride + kw - pad;
            } else {     // lhs_dilation=2, pad=2
                const int th = goh + kh - 2;
                const int tw = gow + kw - 2;
                ok = ok && th >= 0 && tw >= 0 && !(th & 1) && !(tw & 1);
                ih = th >> 1;
                iw = tw >> 1;
            }
            ok = ok && ((unsigned)ih < (unsigned)H) && ((unsigned)iw < (unsigned)W);
            const long long a = ok ? (((long long)(gb * Cin + ci) * H + ih) * W + iw)
                                   : 0ll;
            float xv = X[a];                               // always in-bounds
            xv = ok ? xv : 0.f;
            if (preRelu) xv = fmaxf(xv, 0.f);
            base[idx] = (_Float16)xv;
        }
    };

    v8f acc00 = {};   // a0 x b(sub0)
    v8f acc01 = {};   // a0 x b(sub1)
    v8f acc10 = {};   // a1 x b(sub0)
    v8f acc11 = {};   // a1 x b(sub1)

    gather_step(0, &stage[0][0][0]);
    __syncthreads();

    int p = 0;
    for (int k0 = 0; k0 < Kpad; k0 += 32) {
        // B fragments: 32 contiguous bytes per lane per sub-tile from LDS.
        const _Float16* sp0 = &stage[p][0][lane << 4];
        const v8h b0lo = *(const v8h*)sp0;
        const v8h b0hi = *(const v8h*)(sp0 + 8);
        const v16h bf0 = __builtin_shufflevector(b0lo, b0hi,
                             0,1,2,3,4,5,6,7,8,9,10,11,12,13,14,15);
        const _Float16* sp1 = &stage[p][1][lane << 4];
        const v8h b1lo = *(const v8h*)sp1;
        const v8h b1hi = *(const v8h*)(sp1 + 8);
        const v16h bf1 = __builtin_shufflevector(b1lo, b1hi,
                             0,1,2,3,4,5,6,7,8,9,10,11,12,13,14,15);

        // A fragments: two contiguous b128 loads per 16-row tile.
        const _Float16* wp0 = WF + (size_t)(mbase + l16) * Kpad + k0 + (half << 3);
        const v8h a0lo = *(const v8h*)wp0;
        const v8h a0hi = *(const v8h*)(wp0 + 16);
        const v16h a0 = __builtin_shufflevector(a0lo, a0hi,
                            0,1,2,3,4,5,6,7,8,9,10,11,12,13,14,15);
        const _Float16* wp1 = wp0 + (size_t)16 * Kpad;
        const v8h a1lo = *(const v8h*)wp1;
        const v8h a1hi = *(const v8h*)(wp1 + 16);
        const v16h a1 = __builtin_shufflevector(a1lo, a1hi,
                            0,1,2,3,4,5,6,7,8,9,10,11,12,13,14,15);

        // Overlap: gather next chunk while WMMAs issue.
        if (k0 + 32 < Kpad) gather_step(k0 + 32, &stage[p ^ 1][0][0]);

        acc00 = __builtin_amdgcn_wmma_f32_16x16x32_f16(false, a0, false, bf0,
                                                       (short)0, acc00, false, false);
        acc10 = __builtin_amdgcn_wmma_f32_16x16x32_f16(false, a1, false, bf0,
                                                       (short)0, acc10, false, false);
        acc01 = __builtin_amdgcn_wmma_f32_16x16x32_f16(false, a0, false, bf1,
                                                       (short)0, acc01, false, false);
        acc11 = __builtin_amdgcn_wmma_f32_16x16x32_f16(false, a1, false, bf1,
                                                       (short)0, acc11, false, false);
        __syncthreads();
        p ^= 1;
    }

    // Epilogue: bias, residual, tanh, store. C/D layout: M = r + 8*half,
    // N = lane & 15; two column sets (sub-tiles).
#pragma unroll
    for (int s = 0; s < 2; ++s) {
        const int ncol = nbase + (s << 4) + l16;
        const int bcol = ncol / OHW;
        const int remc = ncol - bcol * OHW;
        const int ohc  = remc / OW;
        const int owc  = remc - ohc * OW;
        const long long obase = ((long long)bcol * Cout) * OHW
                              + (long long)ohc * OW + owc;
        const v8f aLo = s ? acc01 : acc00;
        const v8f aHi = s ? acc11 : acc10;
#pragma unroll
        for (int r = 0; r < 8; ++r) {
            const int mA = mbase + (half << 3) + r;
            const int mB = mA + 16;
            if (mA < Cout) {
                float val = aLo[r] + Bias[mA];
                const long long o = obase + (long long)mA * OHW;
                if (Res) val += Res[o];
                if (doTanh) val = tanhf(val);
                Y[o] = val;
            }
            if (mB < Cout) {
                float val = aHi[r] + Bias[mB];
                const long long o = obase + (long long)mB * OHW;
                if (Res) val += Res[o];
                if (doTanh) val = tanhf(val);
                Y[o] = val;
            }
        }
    }
}

// ---------------------------------------------------------------------------
// Weight pre-conversion: f32 OIHW -> f16 [Mpad][Kpad], zero-padded.
// ---------------------------------------------------------------------------
__global__ __launch_bounds__(256)
void convert_weights(const float* __restrict__ Wt, _Float16* __restrict__ WF,
                     int Cout, int Ktot, int Mpad, int Kpad)
{
    const int i = blockIdx.x * 256 + threadIdx.x;
    if (i >= Mpad * Kpad) return;
    const int m = i / Kpad;
    const int k = i - m * Kpad;
    const float v = (m < Cout && k < Ktot) ? Wt[m * Ktot + k] : 0.f;
    WF[i] = (_Float16)v;
}

// ---------------------------------------------------------------------------
// VQ: nearest-code assignment. One wave per latent vector (N=1024, C=128,
// K=512 codes). z_e layout NCHW [4,128,16,16]; vector n = (b,hh,ww).
// ---------------------------------------------------------------------------
__global__ __launch_bounds__(32)
void vq_assign(const float* __restrict__ zE, const float* __restrict__ emb,
               int* __restrict__ idxOut, int* __restrict__ counts,
               float* __restrict__ outIndices)
{
    __shared__ float zv[128];
    const int n   = blockIdx.x;          // 0..1023
    const int b   = n >> 8;
    const int hh  = (n >> 4) & 15;
    const int ww  = n & 15;
    const int lane = threadIdx.x;

    for (int d = lane; d < 128; d += 32)
        zv[d] = zE[((b * 128 + d) * 16 + hh) * 16 + ww];
    __syncthreads();

    float best = 3.4e38f;
    int bestc = 0x7fffffff;
    for (int c = lane; c < 512; c += 32) {
        const float* e = emb + c * 128;
        float acc = 0.f;
#pragma unroll 4
        for (int d = 0; d < 128; ++d) {
            const float df = zv[d] - e[d];
            acc = fmaf(df, df, acc);
        }
        if (acc < best) { best = acc; bestc = c; }   // strict < keeps lowest c
    }
    for (int off = 16; off > 0; off >>= 1) {
        const float ob = __shfl_down(best, off, 32);
        const int   oc = __shfl_down(bestc, off, 32);
        if (ob < best || (ob == best && oc < bestc)) { best = ob; bestc = oc; }
    }
    if (lane == 0) {
        idxOut[n] = bestc;
        outIndices[n] = (float)bestc;
        atomicAdd(&counts[bestc], 1);
    }
}

// ---------------------------------------------------------------------------
// Gather z_q (decoder input, NCHW) and accumulate sum((z_e - z_q)^2).
// ---------------------------------------------------------------------------
__global__ __launch_bounds__(256)
void vq_gather_mse(const float* __restrict__ zE, const float* __restrict__ emb,
                   const int* __restrict__ idx, float* __restrict__ zQ,
                   float* __restrict__ mseAcc)
{
    __shared__ float sred[256];
    const int i = blockIdx.x * 256 + threadIdx.x;      // over 4*128*16*16
    float d2 = 0.f;
    if (i < 4 * 128 * 16 * 16) {
        const int ww = i & 15;
        int t = i >> 4;
        const int hh = t & 15; t >>= 4;
        const int c  = t & 127;
        const int b  = t >> 7;
        const int n  = (b << 8) + (hh << 4) + ww;
        const float q = emb[idx[n] * 128 + c];
        zQ[i] = q;
        const float d = zE[i] - q;
        d2 = d * d;
    }
    sred[threadIdx.x] = d2;
    __syncthreads();
    for (int s = 128; s > 0; s >>= 1) {
        if (threadIdx.x < s) sred[threadIdx.x] += sred[threadIdx.x + s];
        __syncthreads();
    }
    if (threadIdx.x == 0) atomicAdd(mseAcc, sred[0]);
}

__global__ __launch_bounds__(512)
void vq_finalize(const int* __restrict__ counts, const float* __restrict__ mseAcc,
                 float* __restrict__ outLoss, float* __restrict__ outPerp)
{
    __shared__ float sred[512];
    const int t = threadIdx.x;
    const float p  = (float)counts[t] * (1.0f / 1024.0f);
    const float pc = fmaxf(p, 1e-10f);                 // jnp.clip(p, 1e-10)
    sred[t] = p * logf(pc);
    __syncthreads();
    for (int s = 256; s > 0; s >>= 1) {
        if (t < s) sred[t] += sred[t + s];
        __syncthreads();
    }
    if (t == 0) {
        *outPerp = expf(-sred[0]);
        const float mse = *mseAcc * (1.0f / 131072.0f);
        *outLoss = 1.25f * mse;                        // commit(0.25) + codebook
    }
}

__global__ void vq_zero(int* __restrict__ counts, float* __restrict__ mseAcc)
{
    const int t = blockIdx.x * blockDim.x + threadIdx.x;
    if (t < 512) counts[t] = 0;
    if (t == 512) *mseAcc = 0.f;
}

// ---------------------------------------------------------------------------
// Host-side orchestration
// ---------------------------------------------------------------------------
static inline void launch_conv(hipStream_t s,
                               const float* X, const float* Wt, const float* Bi,
                               const float* Res, float* Y, _Float16* WFbuf,
                               int Nb, int Cin, int H, int W, int Cout,
                               int KH, int KW, int stride, int pad,
                               int mode, int preRelu, int doTanh)
{
    int OH, OW;
    if (mode == 0) {
        OH = (H + 2 * pad - KH) / stride + 1;
        OW = (W + 2 * pad - KW) / stride + 1;
    } else {              // ConvTranspose2d(k=4, s=2, p=1): out = 2*in
        OH = 2 * H;
        OW = 2 * W;
    }
    const int Ktot = Cin * KH * KW;
    const int Kpad = (Ktot + 31) & ~31;
    const int mtiles = (Cout + 31) >> 5;
    const int Mpad = mtiles << 5;

    // Pre-convert weights (stream-ordered; WFbuf reused across layers).
    const int welems = Mpad * Kpad;
    hipLaunchKernelGGL(convert_weights, dim3((welems + 255) / 256), dim3(256),
                       0, s, Wt, WFbuf, Cout, Ktot, Mpad, Kpad);

    // Waves per block: largest of {4,3,2,1} dividing mtiles (no idle waves).
    int wpb = 1;
    if      (mtiles % 4 == 0) wpb = 4;
    else if (mtiles % 3 == 0) wpb = 3;
    else if (mtiles % 2 == 0) wpb = 2;
    const int mgroups = mtiles / wpb;
    const int ntiles  = (Nb * OH * OW) >> 5;   // 32 positions per block

    dim3 grid(mgroups * ntiles), block(32 * wpb);
    hipLaunchKernelGGL(conv_igemm_wmma, grid, block, 0, s,
                       X, WFbuf, Bi, Res, Y, Cin, H, W, Cout,
                       OH, OW, KH, KW, stride, pad, mode, preRelu, doTanh, Kpad);
}

extern "C" void kernel_launch(void* const* d_in, const int* in_sizes, int n_in,
                              void* d_out, int out_size, void* d_ws, size_t ws_size,
                              hipStream_t stream)
{
    (void)in_sizes; (void)n_in; (void)out_size; (void)ws_size;
    // Input order: x, then params in insertion order (w,b pairs), then emb.
    const float* x = (const float*)d_in[0];
    #define PW(i) ((const float*)d_in[i])
    const float* emb = PW(41);

    // Workspace carving, 256B aligned chunks.
    char* wsb = (char*)d_ws;
    size_t off = 0;
    auto carveB = [&](size_t bytes) -> void* {
        void* p = (void*)(wsb + off);
        off += (bytes + 255) & ~(size_t)255;
        return p;
    };
    float* A    = (float*)carveB(6291456 * 4);   // max 4x96x128x128
    float* Bb   = (float*)carveB(3145728 * 4);   // max 4x192x64x64
    float* Cb   = (float*)carveB(1572864 * 4);   // max 4x384x32x32
    float* zE   = (float*)carveB(131072 * 4);    // 4x128x16x16
    float* zQ   = (float*)carveB(131072 * 4);
    float* mse  = (float*)carveB(4);
    int*   idx  = (int*)carveB(1024 * 4);
    int*   cnts = (int*)carveB(512 * 4);
    _Float16* WF = (_Float16*)carveB((size_t)768 * 6912 * 2);  // max f16 weights

    float* recon   = (float*)d_out;            // 4*3*256*256 = 786432
    float* outLoss = recon + 786432;
    float* outPerp = recon + 786433;
    float* outIdx  = recon + 786434;           // 1024 entries

    // ---------------- encoder ----------------
    launch_conv(stream, x,  PW(1),  PW(2),  nullptr, A,  WF, 4,   3, 256, 256,  96, 4,4, 2,1, 0, 0,0);
    launch_conv(stream, A,  PW(3),  PW(4),  nullptr, Bb, WF, 4,  96, 128, 128, 192, 4,4, 2,1, 0, 1,0);
    launch_conv(stream, Bb, PW(5),  PW(6),  nullptr, Cb, WF, 4, 192,  64,  64, 384, 4,4, 2,1, 0, 1,0);
    launch_conv(stream, Cb, PW(7),  PW(8),  nullptr, A,  WF, 4, 384,  32,  32, 768, 4,4, 2,1, 0, 1,0);
    // residual block 1: Cb = A + conv(relu(conv(relu(A))))
    launch_conv(stream, A,  PW(9),  PW(10), nullptr, Bb, WF, 4, 768, 16, 16, 768, 3,3, 1,1, 0, 1,0);
    launch_conv(stream, Bb, PW(11), PW(12), A,       Cb, WF, 4, 768, 16, 16, 768, 1,1, 1,0, 0, 1,0);
    // residual block 2: A = Cb + ...
    launch_conv(stream, Cb, PW(13), PW(14), nullptr, Bb, WF, 4, 768, 16, 16, 768, 3,3, 1,1, 0, 1,0);
    launch_conv(stream, Bb, PW(15), PW(16), Cb,      A,  WF, 4, 768, 16, 16, 768, 1,1, 1,0, 0, 1,0);
    launch_conv(stream, A,  PW(17), PW(18), nullptr, Bb, WF, 4, 768, 16, 16, 192, 3,3, 1,1, 0, 1,0);
    launch_conv(stream, Bb, PW(19), PW(20), nullptr, zE, WF, 4, 192, 16, 16, 128, 1,1, 1,0, 0, 0,0);

    // ---------------- vector quantization ----------------
    hipLaunchKernelGGL(vq_zero, dim3(3), dim3(256), 0, stream, cnts, mse);
    hipLaunchKernelGGL(vq_assign, dim3(1024), dim3(32), 0, stream,
                       zE, emb, idx, cnts, outIdx);
    hipLaunchKernelGGL(vq_gather_mse, dim3(512), dim3(256), 0, stream,
                       zE, emb, idx, zQ, mse);
    hipLaunchKernelGGL(vq_finalize, dim3(1), dim3(512), 0, stream,
                       cnts, mse, outLoss, outPerp);

    // ---------------- decoder (forward value of z_q_st == z_q) ----------------
    launch_conv(stream, zQ, PW(21), PW(22), nullptr, A,  WF, 4, 128, 16, 16, 192, 1,1, 1,0, 0, 0,0);
    launch_conv(stream, A,  PW(23), PW(24), nullptr, Bb, WF, 4, 192, 16, 16, 768, 3,3, 1,1, 0, 0,0);
    // residual block 1: A = Bb + ...
    launch_conv(stream, Bb, PW(25), PW(26), nullptr, Cb, WF, 4, 768, 16, 16, 768, 3,3, 1,1, 0, 1,0);
    launch_conv(stream, Cb, PW(27), PW(28), Bb,      A,  WF, 4, 768, 16, 16, 768, 1,1, 1,0, 0, 1,0);
    // residual block 2: Bb = A + ...
    launch_conv(stream, A,  PW(29), PW(30), nullptr, Cb, WF, 4, 768, 16, 16, 768, 3,3, 1,1, 0, 1,0);
    launch_conv(stream, Cb, PW(31), PW(32), A,       Bb, WF, 4, 768, 16, 16, 768, 1,1, 1,0, 0, 1,0);
    // upsampling deconvs (mode 1), relu on input, tanh on last
    launch_conv(stream, Bb, PW(33), PW(34), nullptr, A,     WF, 4, 768,  16,  16, 384, 4,4, 1,2, 1, 1,0);
    launch_conv(stream, A,  PW(35), PW(36), nullptr, Bb,    WF, 4, 384,  32,  32, 192, 4,4, 1,2, 1, 1,0);
    launch_conv(stream, Bb, PW(37), PW(38), nullptr, A,     WF, 4, 192,  64,  64,  96, 4,4, 1,2, 1, 1,0);
    launch_conv(stream, A,  PW(39), PW(40), nullptr, recon, WF, 4,  96, 128, 128,   3, 4,4, 1,2, 1, 1,1);
    #undef PW
}
